// MoE_13159779794954
// MI455X (gfx1250) — compile-verified, matrix operand order checked by reading, not run on hardware
//
#include <hip/hip_runtime.h>
#include <math.h>

// ---------------- problem constants ----------------
#define E_    8
#define M_    1024
#define H_    4096
#define TOK_  16384          // B*S
#define CAP_  2048           // ceil(TOK/E * 1.0)
#define OUTN_ (TOK_*M_)      // flat elements of main output

typedef unsigned short u16;
typedef __attribute__((ext_vector_type(16))) __bf16 v16bf;
typedef __attribute__((ext_vector_type(8)))  float  v8f;

union Frag { uint4 q[2]; v16bf v; };   // 32 bytes: 16 bf16 halves = 8 VGPRs

// ---- CDNA5 async global->LDS path (guarded; falls back to register staging) ----
#if __has_builtin(__builtin_amdgcn_global_load_async_to_lds_b128)
#define ASYNC_LDS 1
// Diagnostic-confirmed param 0 type: 'int __attribute__((vector_size(16))) __device__ *'
typedef int i32x4 __attribute__((vector_size(16)));
typedef __attribute__((address_space(1))) i32x4 gv_t;   // global (= __device__)
typedef __attribute__((address_space(3))) i32x4 lv_t;   // LDS
#define ASYNC_B128(g, l, off) \
  __builtin_amdgcn_global_load_async_to_lds_b128((gv_t*)(g), (lv_t*)(l), (off), 0)
#else
#define ASYNC_LDS 0
#endif

#if __has_builtin(__builtin_amdgcn_s_wait_asynccnt)
#define WAIT_ASYNC(n) __builtin_amdgcn_s_wait_asynccnt(n)
#else
#define WAIT_ASYNC(n) asm volatile("s_wait_asynccnt %0" :: "i"(n) : "memory")
#endif

__device__ __forceinline__ u16 f2bf(float f) {           // round-to-nearest-even
  unsigned u = __float_as_uint(f);
  unsigned r = 0x7FFFu + ((u >> 16) & 1u);
  return (u16)((u + r) >> 16);
}
__device__ __forceinline__ float gelu_t(float x) {       // tanh-approx gelu (jax default)
  return 0.5f * x * (1.0f + tanhf(0.7978845608028654f * (x + 0.044715f * x * x * x)));
}

// ---------------- 1) gating: logits, softmax, argmax ----------------
// one wave32 per token; deterministic per-block me partials (no float atomics)
__global__ __launch_bounds__(256) void gate_kernel(
    const float* __restrict__ x, const float* __restrict__ wg,
    int* __restrict__ idx, float* __restrict__ gate, float* __restrict__ me_partial) {
  __shared__ float sMe[8][8];
  int lane = threadIdx.x & 31, wid = threadIdx.x >> 5;
  int t = blockIdx.x * 8 + wid;
  float acc[8] = {};
  const float* xr = x + (size_t)t * M_;
  for (int m = lane; m < M_; m += 32) {
    float xv = xr[m];
    const float* wr = wg + m * 8;
#pragma unroll
    for (int e = 0; e < 8; e++) acc[e] += xv * wr[e];
  }
#pragma unroll
  for (int off = 16; off; off >>= 1)
#pragma unroll
    for (int e = 0; e < 8; e++) acc[e] += __shfl_xor(acc[e], off, 32);
  if (lane == 0) {
    float mx = acc[0]; int im = 0;
#pragma unroll
    for (int e = 1; e < 8; e++) if (acc[e] > mx) { mx = acc[e]; im = e; }
    float ex[8], s = 0.f;
#pragma unroll
    for (int e = 0; e < 8; e++) { ex[e] = __expf(acc[e] - mx); s += ex[e]; }
    float inv = 1.f / s;
    idx[t] = im;
    gate[t] = inv;                 // exp(max-max)/sum == 1/sum
#pragma unroll
    for (int e = 0; e < 8; e++) sMe[wid][e] = ex[e] * inv;
  }
  __syncthreads();
  if (threadIdx.x < 8) {
    float s = 0.f;
#pragma unroll
    for (int w = 0; w < 8; w++) s += sMe[w][threadIdx.x];
    me_partial[blockIdx.x * 8 + threadIdx.x] = s;
  }
}

// ---------------- 2) per-chunk expert histograms (chunk = 256 tokens) ----------------
__global__ __launch_bounds__(256) void hist_kernel(const int* __restrict__ idx,
                                                   int* __restrict__ chunk_hist) {
  __shared__ int h[8];
  if (threadIdx.x < 8) h[threadIdx.x] = 0;
  __syncthreads();
  atomicAdd(&h[idx[blockIdx.x * 256 + threadIdx.x]], 1);
  __syncthreads();
  if (threadIdx.x < 8) chunk_hist[blockIdx.x * 8 + threadIdx.x] = h[threadIdx.x];
}

// ---------------- 3) serial chunk scan + l_aux + exp_counts ----------------
__global__ void scan_kernel(const int* __restrict__ chunk_hist, int* __restrict__ chunk_base,
                            int* __restrict__ counts, const float* __restrict__ me_partial,
                            float* __restrict__ dout) {
  __shared__ float sMe[8];
  __shared__ int   sC[8];
  int e = threadIdx.x;
  if (e < 8) {
    int run = 0;
    for (int c = 0; c < 64; c++) { chunk_base[c * 8 + e] = run; run += chunk_hist[c * 8 + e]; }
    counts[e] = run; sC[e] = run;
    float ms = 0.f;
    for (int b = 0; b < TOK_ / 8; b++) ms += me_partial[b * 8 + e];
    sMe[e] = ms;
  }
  __syncthreads();
  if (e == 0) {
    float la = 0.f, invT = 1.f / (float)TOK_;
    for (int k = 0; k < 8; k++) la += (sMe[k] * invT) * ((float)sC[k] * invT);
    dout[OUTN_] = la * (float)E_;
    int* ic = (int*)dout;                       // exp_counts stored as int32 bit patterns
    for (int k = 0; k < 8; k++) ic[OUTN_ + 1 + k] = sC[k];
  }
}

// ---------------- 4) ordered slot assignment via wave32 ballots ----------------
__global__ __launch_bounds__(256) void loc_kernel(const int* __restrict__ idx,
                                                  const int* __restrict__ chunk_base,
                                                  int* __restrict__ slot2token) {
  __shared__ int sCnt[8][8];
  int lane = threadIdx.x & 31, wid = threadIdx.x >> 5;
  int t = blockIdx.x * 256 + threadIdx.x;
  int my = idx[t];
  unsigned m[8];
#pragma unroll
  for (int e = 0; e < 8; e++) m[e] = (unsigned)__ballot(my == e);
  if (lane < 8) sCnt[wid][lane] = __popc(m[lane]);
  __syncthreads();
  int base = 0;
  for (int w = 0; w < wid; w++) base += sCnt[w][my];
  int rank = __popc(m[my] & ((1u << lane) - 1u));
  int loc = chunk_base[blockIdx.x * 8 + my] + base + rank;
  if (loc < CAP_) slot2token[my * CAP_ + loc] = t;   // dropped tokens never stored
}

// ---------------- 5) dispatch: gather kept token rows -> bf16 [E][CAP][M] ----------------
__global__ __launch_bounds__(256) void dispatch_kernel(const float* __restrict__ x,
                                                       const int* __restrict__ slot2token,
                                                       u16* __restrict__ disp) {
  int lane = threadIdx.x & 31, wid = threadIdx.x >> 5;
  int slot = blockIdx.x * 8 + wid;
  int tok = slot2token[slot];
  if (tok < 0) return;                                // empty slot stays memset-zero
  const float4* src = (const float4*)(x + (size_t)tok * M_);
  ushort4* dst = (ushort4*)(disp + (size_t)slot * M_);
  for (int i = lane; i < M_ / 4; i += 32) {
    float4 v = src[i];
    ushort4 o; o.x = f2bf(v.x); o.y = f2bf(v.y); o.z = f2bf(v.z); o.w = f2bf(v.w);
    dst[i] = o;
  }
}

// ---------------- 6) weight convert + transpose: f32 [E][R][C] -> bf16 [E][C][R] ----------------
__global__ __launch_bounds__(256) void convT_kernel(const float* __restrict__ w,
                                                    u16* __restrict__ wt, int R, int C) {
  __shared__ float s[32][33];
  int e = blockIdx.z;
  const float* in = w + (size_t)e * R * C;
  u16* out = wt + (size_t)e * R * C;
  int tx = threadIdx.x & 31, ty = threadIdx.x >> 5;
  int r0 = blockIdx.y * 32, c0 = blockIdx.x * 32;
#pragma unroll
  for (int j = 0; j < 4; j++) s[ty + 8 * j][tx] = in[(size_t)(r0 + ty + 8 * j) * C + c0 + tx];
  __syncthreads();
#pragma unroll
  for (int j = 0; j < 4; j++) out[(size_t)(c0 + ty + 8 * j) * R + r0 + tx] = f2bf(s[tx][ty + 8 * j]);
}

// ---------------- WMMA compute step for one 32-deep K slice ----------------
__device__ __forceinline__ void wmma_step(const u16* cA, const u16* cB,
                                          int wr, int wc, int rsel, int rl,
                                          v8f acc[4][2]) {
  const int LD = 40;
  Frag af[4], bf_[2];
#pragma unroll
  for (int mf = 0; mf < 4; mf++) {
    const u16* p = cA + (wr * 64 + mf * 16 + rl) * LD + rsel;
    af[mf].q[0] = *(const uint4*)p;
    af[mf].q[1] = *(const uint4*)(p + 16);
  }
#pragma unroll
  for (int nf = 0; nf < 2; nf++) {
    const u16* p = cB + (wc * 32 + nf * 16 + rl) * LD + rsel;
    bf_[nf].q[0] = *(const uint4*)p;
    bf_[nf].q[1] = *(const uint4*)(p + 16);
  }
#pragma unroll
  for (int mf = 0; mf < 4; mf++)
#pragma unroll
    for (int nf = 0; nf < 2; nf++)
      acc[mf][nf] = __builtin_amdgcn_wmma_f32_16x16x32_bf16(
          false, af[mf].v, false, bf_[nf].v, (short)0, acc[mf][nf], false, false);
}

// ---------------- shared GEMM mainloop: 128x128 tile, BK=32, bf16 WMMA ----------------
// A: row-major [rows][K] bf16; B: K-contiguous per output column (pre-transposed) [cols][K].
// 8 waves, each owns a 64x32 sub-tile = 4x2 fragments of v_wmma_f32_16x16x32_bf16.
// Async path: double-buffered LDS, GLOBAL_LOAD_ASYNC_TO_LDS_B128 overlaps tile i+1 copy
// with tile i compute; s_wait_asynccnt 4 retires the older tile's 4 loads per thread.
__device__ __forceinline__ void gemm_tile(const u16* __restrict__ Arow,
                                          const u16* __restrict__ Brow, int K,
                                          u16* sA, u16* sB, v8f acc[4][2]) {
  const int LD = 40;                                   // padded halves per row (80B)
  const int BUF = 128 * LD;                            // halves per LDS buffer
  int t = threadIdx.x;
  int lane = t & 31, wid = t >> 5;
  int row = t >> 1, hoff = (t & 1) * 16;               // 256 thr cover 128 rows x 32 halves
  int wr = wid >> 2, wc = wid & 3;                     // wave 2x4 grid over 128x128
  int rsel = (lane >> 4) << 3;                         // 16-bit A layout: lanes>=16 -> K+8
  int rl = lane & 15;
  int nk = K >> 5;
#if ASYNC_LDS
  const u16* gA = Arow + (size_t)row * K + hoff;
  const u16* gB = Brow + (size_t)row * K + hoff;
  u16* lA = sA + row * LD + hoff;
  u16* lB = sB + row * LD + hoff;
  // ISA: INST_OFFSET is added to BOTH global and LDS address, so the second 16B
  // chunk of each 32B strip is just offset:16 on the same address pair.
  ASYNC_B128(gA, lA, 0);
  ASYNC_B128(gA, lA, 16);
  ASYNC_B128(gB, lB, 0);
  ASYNC_B128(gB, lB, 16);
  for (int ks = 0; ks < nk; ++ks) {
    int kb = ks << 5;
    if (ks + 1 < nk) {
      int nb = (ks + 1) & 1;
      ASYNC_B128(gA + kb + 32, lA + nb * BUF, 0);
      ASYNC_B128(gA + kb + 32, lA + nb * BUF, 16);
      ASYNC_B128(gB + kb + 32, lB + nb * BUF, 0);
      ASYNC_B128(gB + kb + 32, lB + nb * BUF, 16);
      __builtin_prefetch((const void*)(gA + kb + 96), 0, 1);     // 2 tiles ahead -> GL2
      __builtin_prefetch((const void*)(gB + kb + 96), 0, 1);
      WAIT_ASYNC(4);                      // older tile's 4 loads done; next 4 in flight
    } else {
      WAIT_ASYNC(0);
    }
    __syncthreads();                      // all waves' async data visible in LDS
    wmma_step(sA + (ks & 1) * BUF, sB + (ks & 1) * BUF, wr, wc, rsel, rl, acc);
    __syncthreads();                      // reads done before this buffer is re-filled
  }
#else
  for (int kb = 0; kb < K; kb += 32) {
    const uint4* pa = (const uint4*)(Arow + (size_t)row * K + kb + hoff);
    const uint4* pb = (const uint4*)(Brow + (size_t)row * K + kb + hoff);
    uint4 a0 = pa[0], a1 = pa[1], b0 = pb[0], b1 = pb[1];
    __builtin_prefetch((const void*)(Arow + (size_t)row * K + kb + 64 + hoff), 0, 1);
    __builtin_prefetch((const void*)(Brow + (size_t)row * K + kb + 64 + hoff), 0, 1);
    __syncthreads();
    *(uint4*)(sA + row * LD + hoff) = a0; *(uint4*)(sA + row * LD + hoff + 8) = a1;
    *(uint4*)(sB + row * LD + hoff) = b0; *(uint4*)(sB + row * LD + hoff + 8) = b1;
    __syncthreads();
    wmma_step(sA, sB, wr, wc, rsel, rl, acc);
  }
#endif
}

// ---------------- 7) GEMM1: h = gelu(disp @ w1 + b1), bf16 out ----------------
__global__ __launch_bounds__(256) void gemm1_kernel(const u16* __restrict__ disp,
                                                    const u16* __restrict__ w1t,
                                                    const float* __restrict__ b1,
                                                    u16* __restrict__ h) {
  __shared__ __align__(16) u16 sA[2 * 128 * 40];
  __shared__ __align__(16) u16 sB[2 * 128 * 40];
  int e = blockIdx.z, m0 = blockIdx.y * 128, n0 = blockIdx.x * 128;
  const u16* Arow = disp + ((size_t)e * CAP_ + m0) * M_;
  const u16* Brow = w1t + ((size_t)e * H_ + n0) * M_;
  v8f acc[4][2] = {};
  gemm_tile(Arow, Brow, M_, sA, sB, acc);
  int lane = threadIdx.x & 31, wid = threadIdx.x >> 5;
  int wr = wid >> 2, wc = wid & 3, rsel = (lane >> 4) << 3, rl = lane & 15;
  const float* b1e = b1 + (size_t)e * H_;
  u16* he = h + (size_t)e * CAP_ * H_;
#pragma unroll
  for (int mf = 0; mf < 4; mf++)
#pragma unroll
    for (int nf = 0; nf < 2; nf++) {
      int gn = n0 + wc * 32 + nf * 16 + rl;
      float bv = b1e[gn];
#pragma unroll
      for (int r = 0; r < 8; r++) {
        int gm = m0 + wr * 64 + mf * 16 + r + rsel;     // C/D layout: VGPR r -> M=r / r+8
        he[(size_t)gm * H_ + gn] = f2bf(gelu_t(acc[mf][nf][r] + bv));
      }
    }
}

// ---------------- 8) GEMM2 + fused combine: out[token] = (h @ w2 + b2) * gate ----------------
__global__ __launch_bounds__(256) void gemm2_kernel(const u16* __restrict__ h,
                                                    const u16* __restrict__ w2t,
                                                    const float* __restrict__ b2,
                                                    const int* __restrict__ slot2token,
                                                    const float* __restrict__ gate,
                                                    float* __restrict__ out) {
  __shared__ __align__(16) u16 sA[2 * 128 * 40];
  __shared__ __align__(16) u16 sB[2 * 128 * 40];
  __shared__ int   sTok[128];
  __shared__ float sG[128];
  int e = blockIdx.z, m0 = blockIdx.y * 128, n0 = blockIdx.x * 128;
  const u16* Arow = h + ((size_t)e * CAP_ + m0) * H_;
  const u16* Brow = w2t + ((size_t)e * M_ + n0) * H_;
  v8f acc[4][2] = {};
  gemm_tile(Arow, Brow, H_, sA, sB, acc);
  __syncthreads();
  if (threadIdx.x < 128) {
    int tok = slot2token[(size_t)e * CAP_ + m0 + threadIdx.x];
    sTok[threadIdx.x] = tok;
    sG[threadIdx.x] = (tok >= 0) ? gate[tok] : 0.f;
  }
  __syncthreads();
  int lane = threadIdx.x & 31, wid = threadIdx.x >> 5;
  int wr = wid >> 2, wc = wid & 3, rsel = (lane >> 4) << 3, rl = lane & 15;
  const float* b2e = b2 + (size_t)e * M_;
#pragma unroll
  for (int mf = 0; mf < 4; mf++)
#pragma unroll
    for (int nf = 0; nf < 2; nf++) {
      int gn = n0 + wc * 32 + nf * 16 + rl;
      float bv = b2e[gn];
#pragma unroll
      for (int r = 0; r < 8; r++) {
        int lr = wr * 64 + mf * 16 + r + rsel;
        int tok = sTok[lr];
        if (tok >= 0) out[(size_t)tok * M_ + gn] = (acc[mf][nf][r] + bv) * sG[lr];
      }
    }
}

// ---------------- host: orchestrate on `stream` (graph-capture safe) ----------------
extern "C" void kernel_launch(void* const* d_in, const int* in_sizes, int n_in,
                              void* d_out, int out_size, void* d_ws, size_t ws_size,
                              hipStream_t stream) {
  (void)in_sizes; (void)n_in; (void)ws_size;
  const float* x  = (const float*)d_in[0];
  const float* wg = (const float*)d_in[1];
  const float* w1 = (const float*)d_in[2];
  const float* b1 = (const float*)d_in[3];
  const float* w2 = (const float*)d_in[4];
  const float* b2 = (const float*)d_in[5];
  float* out = (float*)d_out;

  char* ws = (char*)d_ws;
  size_t off = 0;
  auto take = [&](size_t bytes) -> char* {
    char* p = ws + off; off = (off + bytes + 255) & ~(size_t)255; return p;
  };
  int*   idx   = (int*)  take((size_t)TOK_ * 4);
  float* gate  = (float*)take((size_t)TOK_ * 4);
  int*   chist = (int*)  take(64 * 8 * 4);
  int*   cbase = (int*)  take(64 * 8 * 4);
  int*   cnts  = (int*)  take(8 * 4);
  float* mep   = (float*)take((size_t)(TOK_ / 8) * 8 * 4);
  int*   s2t   = (int*)  take((size_t)E_ * CAP_ * 4);
  u16*   w1t   = (u16*)  take((size_t)E_ * M_ * H_ * 2);
  u16*   w2t   = (u16*)  take((size_t)E_ * M_ * H_ * 2);
  u16*   disp  = (u16*)  take((size_t)E_ * CAP_ * M_ * 2);
  u16*   hbuf  = (u16*)  take((size_t)E_ * CAP_ * H_ * 2);

  (void)hipMemsetAsync(d_out, 0, (size_t)out_size * sizeof(float), stream); // dropped rows -> 0
  (void)hipMemsetAsync(s2t, 0xFF, (size_t)E_ * CAP_ * 4, stream);           // -1 = empty slot
  (void)hipMemsetAsync(disp, 0, (size_t)E_ * CAP_ * M_ * 2, stream);        // empty slots -> 0

  gate_kernel<<<TOK_ / 8, 256, 0, stream>>>(x, wg, idx, gate, mep);
  hist_kernel<<<64, 256, 0, stream>>>(idx, chist);
  scan_kernel<<<1, 256, 0, stream>>>(chist, cbase, cnts, mep, out);
  loc_kernel<<<64, 256, 0, stream>>>(idx, cbase, s2t);
  dispatch_kernel<<<E_ * CAP_ / 8, 256, 0, stream>>>(x, s2t, disp);
  convT_kernel<<<dim3(H_ / 32, M_ / 32, E_), 256, 0, stream>>>(w1, w1t, M_, H_);
  convT_kernel<<<dim3(M_ / 32, H_ / 32, E_), 256, 0, stream>>>(w2, w2t, H_, M_);
  gemm1_kernel<<<dim3(H_ / 128, CAP_ / 128, E_), 256, 0, stream>>>(disp, w1t, b1, hbuf);
  gemm2_kernel<<<dim3(M_ / 128, CAP_ / 128, E_), 256, 0, stream>>>(hbuf, w2t, b2, s2t, gate, out);
}
